// QuantizedTokenizer_1743756722770
// MI455X (gfx1250) — compile-verified, matrix operand order checked by reading, not run on hardware
//
#include <hip/hip_runtime.h>
#include <hip/hip_bf16.h>
#include <stdint.h>

// ---------------------------------------------------------------------------
// Problem constants (from reference): B=32, Q=1024, H=1024, HR=512, C=4096
// ---------------------------------------------------------------------------
#define H_SZ   1024
#define HR_SZ  512
#define C_SZ   4096
#define NTOK   32768          // B*Q
#define COMMIT_W 0.25f

// WMMA tile config: 128x128 block tile, 8 waves (wave32), K-step 32
#define TM  128
#define TN  128
#define TK  32
#define LDT 40                // padded LDS row stride (bf16 elems) -> 80B, 16B aligned
#define BUFE (TM * LDT)       // elems per LDS buffer

typedef __attribute__((ext_vector_type(16))) __bf16 v16bf;
typedef __attribute__((ext_vector_type(8)))  float  v8f;
typedef __attribute__((ext_vector_type(8)))  unsigned short ushort8;
typedef __attribute__((ext_vector_type(4)))  int    v4i;

#define AS1 __attribute__((address_space(1)))
#define AS3 __attribute__((address_space(3)))

// CDNA5 async memory->LDS path (GLOBAL_LOAD_ASYNC_TO_LDS_B128, ASYNCcnt).
// Probe round 2 confirmed the builtins exist; params are (v4i AS1*, v4i AS3*,
// imm offset, imm cpol) -- pointee must be the int4 payload type.
#if defined(__has_builtin)
#if __has_builtin(__builtin_amdgcn_global_load_async_to_lds_b128) && \
    __has_builtin(__builtin_amdgcn_s_wait_asynccnt)
#define USE_ASYNC_LDS 1
#endif
#endif
#ifndef USE_ASYNC_LDS
#define USE_ASYNC_LDS 0
#endif

#if USE_ASYNC_LDS
__device__ __forceinline__ void async_b128(const uint16_t* g, uint16_t* l) {
    __builtin_amdgcn_global_load_async_to_lds_b128(
        (AS1 v4i*)(g), (AS3 v4i*)(l), 0, 0);
}
#endif

__device__ __forceinline__ uint16_t f2b(float f) {
    union { float f; uint32_t u; } v; v.f = f;
    uint32_t r = v.u + 0x7FFFu + ((v.u >> 16) & 1u);   // round-to-nearest-even
    return (uint16_t)(r >> 16);
}

// Load one 16x32 bf16 WMMA fragment from an LDS tile (row-major, stride ld).
// ISA layout: lane l (lm=l&15, hi=l>>4) holds row lm, K = hi*8+{0..7} and
// K = 16+hi*8+{0..7} -> two aligned ds_load_b128 per fragment.
__device__ __forceinline__ v16bf load_frag(const uint16_t* p16, int ld) {
    const int lane = threadIdx.x & 31;
    const uint16_t* p = p16 + (lane & 15) * ld + (lane >> 4) * 8;
    union { v16bf v; ushort8 h[2]; } u;
    u.h[0] = *(const ushort8*)(p);
    u.h[1] = *(const ushort8*)(p + 16);
    return u.v;
}

// ---------------------------------------------------------------------------
// Shared MMA pipeline: acc[2][4] (+)= A[bm:bm+128][0:K] * Bt[bn:bn+128][0:K]^T
// Double-buffered LDS, ONE barrier per K-step; next tile's async-to-LDS
// transfers (or global loads) overlap the current tile's WMMAs.
// As/Bs each hold 2 buffers of TM*LDT bf16.
// ---------------------------------------------------------------------------
__device__ __forceinline__ void mma_tiles(
        const uint16_t* __restrict__ A, const uint16_t* __restrict__ Bt,
        int K, int bm, int bn,
        uint16_t* As, uint16_t* Bs, v8f acc[2][4])
{
    const int tid  = threadIdx.x;
    const int wave = tid >> 5;
    const int wm = wave >> 1;          // 0..3 : 32-row slab
    const int wn = wave & 1;           // 0..1 : 64-col slab
    // staging: 512 x 16B chunks per tile, 2 per thread (rows r0 and r0+64)
    const int r0 = tid >> 2;
    const int c0 = (tid & 3) * 8;
    const int NIT = K / TK;

    const uint16_t* gA0 = A  + (size_t)(bm + r0) * K + c0;
    const uint16_t* gA1 = gA0 + (size_t)64 * K;
    const uint16_t* gB0 = Bt + (size_t)(bn + r0) * K + c0;
    const uint16_t* gB1 = gB0 + (size_t)64 * K;
    uint16_t* lA0 = As + r0 * LDT + c0;
    uint16_t* lA1 = lA0 + 64 * LDT;
    uint16_t* lB0 = Bs + r0 * LDT + c0;
    uint16_t* lB1 = lB0 + 64 * LDT;

    #pragma unroll
    for (int mt = 0; mt < 2; ++mt)
        #pragma unroll
        for (int nt = 0; nt < 4; ++nt) { v8f z = {}; acc[mt][nt] = z; }

#if USE_ASYNC_LDS
    // prologue: stage tile 0 into buffer 0 via async DMA to LDS
    async_b128(gA0, lA0);
    async_b128(gA1, lA1);
    async_b128(gB0, lB0);
    async_b128(gB1, lB1);
#else
    ushort8 ra0 = *(const ushort8*)(gA0);
    ushort8 ra1 = *(const ushort8*)(gA1);
    ushort8 rb0 = *(const ushort8*)(gB0);
    ushort8 rb1 = *(const ushort8*)(gB1);
#endif

    for (int it = 0; it < NIT; ++it) {
        const int cur = it & 1;
        const int co  = cur * BUFE;

#if USE_ASYNC_LDS
        __builtin_amdgcn_s_wait_asynccnt(0);   // own chunks of buffer `cur` landed
        __syncthreads();                       // everyone's chunks landed
        if (it + 1 < NIT) {                    // stage next tile into other buffer
            const int kk = (it + 1) * TK;
            const int nb = (cur ^ 1) * BUFE;
            async_b128(gA0 + kk, lA0 + nb);
            async_b128(gA1 + kk, lA1 + nb);
            async_b128(gB0 + kk, lB0 + nb);
            async_b128(gB1 + kk, lB1 + nb);
        }
#else
        *(ushort8*)(lA0 + co) = ra0;           // publish staged tile
        *(ushort8*)(lA1 + co) = ra1;
        *(ushort8*)(lB0 + co) = rb0;
        *(ushort8*)(lB1 + co) = rb1;
        __syncthreads();
        if (it + 1 < NIT) {                    // issue next tile's loads now;
            const int kk = (it + 1) * TK;      // they overlap the WMMAs below
            ra0 = *(const ushort8*)(gA0 + kk);
            ra1 = *(const ushort8*)(gA1 + kk);
            rb0 = *(const ushort8*)(gB0 + kk);
            rb1 = *(const ushort8*)(gB1 + kk);
            // branch-free prefetch of the tile after next (clamped)
            const int kpf = (kk + TK < K) ? (kk + TK) : kk;
            __builtin_prefetch(gA0 + kpf, 0, 1);
        }
#endif
        v16bf af[2], bfv[4];
        #pragma unroll
        for (int mt = 0; mt < 2; ++mt)
            af[mt] = load_frag(As + co + (32 * wm + 16 * mt) * LDT, LDT);
        #pragma unroll
        for (int nt = 0; nt < 4; ++nt)
            bfv[nt] = load_frag(Bs + co + (64 * wn + 16 * nt) * LDT, LDT);
        #pragma unroll
        for (int mt = 0; mt < 2; ++mt)
            #pragma unroll
            for (int nt = 0; nt < 4; ++nt)
                acc[mt][nt] = __builtin_amdgcn_wmma_f32_16x16x32_bf16(
                    false, af[mt], false, bfv[nt], (short)0, acc[mt][nt],
                    false, false);
        // no trailing barrier: re-staging a buffer only happens after the
        // next barrier, which every consumer of that buffer has passed.
    }
}

// ---------------------------------------------------------------------------
// Generic bf16 WMMA GEMM: C[M][N] = A[M][K] * Bt[N][K]^T + bias (+ReLU)
// ---------------------------------------------------------------------------
__global__ __launch_bounds__(256) void gemm_bf16_wmma(
        const uint16_t* __restrict__ A,
        const uint16_t* __restrict__ Bt,
        const float*    __restrict__ bias,
        float*          __restrict__ Cf,
        uint16_t*       __restrict__ Cb,
        int M, int N, int K, int relu)
{
    __shared__ __align__(16) uint16_t As[2 * BUFE];
    __shared__ __align__(16) uint16_t Bs[2 * BUFE];
    const int bm = blockIdx.y * TM;
    const int bn = blockIdx.x * TN;
    const int lane = threadIdx.x & 31;
    const int wave = threadIdx.x >> 5;
    const int wm = wave >> 1, wn = wave & 1;

    v8f acc[2][4];
    mma_tiles(A, Bt, K, bm, bn, As, Bs, acc);

    // Epilogue: C/D layout -> lane (ln,hi) holds col ln, rows r + 8*hi.
    const int hi = lane >> 4;
    const int ln = lane & 15;
    #pragma unroll
    for (int mt = 0; mt < 2; ++mt)
        #pragma unroll
        for (int nt = 0; nt < 4; ++nt) {
            int col = bn + 64 * wn + 16 * nt + ln;
            float bv = bias ? bias[col] : 0.f;
            #pragma unroll
            for (int r = 0; r < 8; ++r) {
                int row = bm + 32 * wm + 16 * mt + 8 * hi + r;
                float v = acc[mt][nt][r] + bv;
                if (relu) v = fmaxf(v, 0.f);
                size_t o = (size_t)row * N + col;
                if (Cf) Cf[o] = v;
                if (Cb) Cb[o] = f2b(v);
            }
        }
}

// ---------------------------------------------------------------------------
// Fused distance GEMM + argmin:  d[n][c] = ||e_c||^2 - 2 * z[n].e_c
// Never materializes the 32768x4096 matrix; keeps running per-row argmin.
// ---------------------------------------------------------------------------
__global__ __launch_bounds__(256) void dist_argmin_wmma(
        const uint16_t* __restrict__ Z,     // [NTOK][HR] bf16
        const uint16_t* __restrict__ CB,    // [C][HR]   bf16 (natural "Bt" layout)
        const float*    __restrict__ cnorm, // [C]
        int*            __restrict__ idx_out,
        float*          __restrict__ tok_out)
{
    __shared__ __align__(16) uint16_t As[2 * BUFE];
    __shared__ __align__(16) uint16_t Bs[2 * BUFE];
    const int bm   = blockIdx.x * TM;
    const int lane = threadIdx.x & 31;
    const int wave = threadIdx.x >> 5;
    const int wm = wave >> 1, wn = wave & 1;
    const int hi = lane >> 4;
    const int ln = lane & 15;

    float rmin[2][8];
    int   ridx[2][8];
    #pragma unroll
    for (int mt = 0; mt < 2; ++mt)
        #pragma unroll
        for (int r = 0; r < 8; ++r) { rmin[mt][r] = 3.4e38f; ridx[mt][r] = 0; }

    for (int cn = 0; cn < C_SZ; cn += TN) {
        v8f acc[2][4];
        mma_tiles(Z, CB, HR_SZ, bm, cn, As, Bs, acc);

        // Running argmin update (ascending cn => first-occurrence semantics)
        #pragma unroll
        for (int mt = 0; mt < 2; ++mt)
            #pragma unroll
            for (int nt = 0; nt < 4; ++nt) {
                int col = cn + 64 * wn + 16 * nt + ln;
                float c2 = cnorm[col];
                #pragma unroll
                for (int r = 0; r < 8; ++r) {
                    float d = c2 - 2.f * acc[mt][nt][r];
                    if (d < rmin[mt][r]) { rmin[mt][r] = d; ridx[mt][r] = col; }
                }
            }
    }

    // Reduce across the 16 lanes (same rows, different cols) of each half-wave.
    #pragma unroll
    for (int mt = 0; mt < 2; ++mt)
        #pragma unroll
        for (int r = 0; r < 8; ++r) {
            float v = rmin[mt][r];
            int  ix = ridx[mt][r];
            #pragma unroll
            for (int s = 8; s >= 1; s >>= 1) {
                float ov = __shfl_xor(v, s, 32);
                int   oi = __shfl_xor(ix, s, 32);
                if (ov < v || (ov == v && oi < ix)) { v = ov; ix = oi; }
            }
            if (ln == 0) {
                int row = bm + 32 * wm + 16 * mt + 8 * hi + r;
                idx_out[row] = ix;
                tok_out[row] = (float)ix;
            }
        }
}

// ---------------------------------------------------------------------------
// Helpers: bf16 convert, transposed weight convert, codebook norms,
// gather + commitment-loss partials, deterministic final reduce.
// ---------------------------------------------------------------------------
__global__ void cvt_f32_bf16(const float* __restrict__ in,
                             uint16_t* __restrict__ out, int n) {
    int i = blockIdx.x * blockDim.x + threadIdx.x;
    if (i < n) out[i] = f2b(in[i]);
}

// in[K][N] fp32 -> out[N][K] bf16 (LDS-tiled transpose)
__global__ __launch_bounds__(256) void transpose_cvt(
        const float* __restrict__ in, uint16_t* __restrict__ out, int K, int N) {
    __shared__ float tile[32][33];
    int n0 = blockIdx.x * 32, k0 = blockIdx.y * 32;
    int tx = threadIdx.x & 31, ty = threadIdx.x >> 5;   // 8 row-groups
    for (int i = ty; i < 32; i += 8)
        tile[i][tx] = in[(size_t)(k0 + i) * N + n0 + tx];
    __syncthreads();
    for (int i = ty; i < 32; i += 8)
        out[(size_t)(n0 + i) * K + k0 + tx] = f2b(tile[tx][i]);
}

__global__ __launch_bounds__(256) void cnorm_kernel(
        const float* __restrict__ cb, float* __restrict__ cnorm) {
    int g = threadIdx.x >> 5, l = threadIdx.x & 31;
    int c = blockIdx.x * 8 + g;
    const float* p = cb + (size_t)c * HR_SZ;
    float s = 0.f;
    for (int k = l; k < HR_SZ; k += 32) { float v = p[k]; s += v * v; }
    #pragma unroll
    for (int sh = 16; sh; sh >>= 1) s += __shfl_xor(s, sh, 32);
    if (l == 0) cnorm[c] = s;
}

__global__ __launch_bounds__(128) void gather_commit(
        const float* __restrict__ z, const float* __restrict__ codebook,
        const int* __restrict__ idx, float* __restrict__ q_out,
        uint16_t* __restrict__ q_b, float* __restrict__ part) {
    int row = blockIdx.x;
    int c = idx[row];
    const float* src = codebook + (size_t)c * HR_SZ;
    const float* zr  = z + (size_t)row * HR_SZ;
    float s = 0.f;
    for (int k = threadIdx.x; k < HR_SZ; k += 128) {
        float q = src[k];
        float d = zr[k] - q;
        s += d * d;
        q_out[(size_t)row * HR_SZ + k] = q;      // quantized_st forward value
        q_b[(size_t)row * HR_SZ + k]  = f2b(q);  // bf16 for decoder GEMM
    }
    #pragma unroll
    for (int sh = 16; sh; sh >>= 1) s += __shfl_xor(s, sh, 32);
    __shared__ float red[4];
    int wave = threadIdx.x >> 5, lane = threadIdx.x & 31;
    if (lane == 0) red[wave] = s;
    __syncthreads();
    if (threadIdx.x == 0) part[row] = red[0] + red[1] + red[2] + red[3];
}

__global__ __launch_bounds__(256) void commit_final(
        const float* __restrict__ part, float* __restrict__ loss) {
    __shared__ float red[8];
    float s = 0.f;
    for (int i = threadIdx.x; i < NTOK; i += 256) s += part[i];
    #pragma unroll
    for (int sh = 16; sh; sh >>= 1) s += __shfl_xor(s, sh, 32);
    int wave = threadIdx.x >> 5, lane = threadIdx.x & 31;
    if (lane == 0) red[wave] = s;
    __syncthreads();
    if (threadIdx.x == 0) {
        float t = 0.f;
        #pragma unroll
        for (int i = 0; i < 8; ++i) t += red[i];
        *loss = COMMIT_W * t / (float)((size_t)NTOK * HR_SZ);
    }
}

// ---------------------------------------------------------------------------
// kernel_launch
// ---------------------------------------------------------------------------
extern "C" void kernel_launch(void* const* d_in, const int* in_sizes, int n_in,
                              void* d_out, int out_size, void* d_ws, size_t ws_size,
                              hipStream_t stream) {
    const float* x        = (const float*)d_in[0];
    const float* enc_w1   = (const float*)d_in[1];
    const float* enc_b1   = (const float*)d_in[2];
    const float* enc_w2   = (const float*)d_in[3];
    const float* enc_b2   = (const float*)d_in[4];
    const float* codebook = (const float*)d_in[5];
    const float* dec_w1   = (const float*)d_in[6];
    const float* dec_b1   = (const float*)d_in[7];
    const float* dec_w2   = (const float*)d_in[8];
    const float* dec_b2   = (const float*)d_in[9];

    // workspace layout (256B aligned slabs)
    char* ws = (char*)d_ws;
    size_t off = 0;
    auto alloc = [&](size_t bytes) -> void* {
        void* p = ws + off;
        off = (off + bytes + 255) & ~(size_t)255;
        return p;
    };
    uint16_t* x_b  = (uint16_t*)alloc((size_t)NTOK * H_SZ  * 2);
    uint16_t* h_b  = (uint16_t*)alloc((size_t)NTOK * H_SZ  * 2);  // enc hidden, reused as dec hidden
    float*    z_f  = (float*)   alloc((size_t)NTOK * HR_SZ * 4);
    uint16_t* z_b  = (uint16_t*)alloc((size_t)NTOK * HR_SZ * 2);
    uint16_t* q_b  = (uint16_t*)alloc((size_t)NTOK * HR_SZ * 2);
    uint16_t* w1t  = (uint16_t*)alloc((size_t)H_SZ  * H_SZ  * 2);
    uint16_t* w2t  = (uint16_t*)alloc((size_t)H_SZ  * HR_SZ * 2);
    uint16_t* w3t  = (uint16_t*)alloc((size_t)HR_SZ * H_SZ  * 2);
    uint16_t* w4t  = (uint16_t*)alloc((size_t)H_SZ  * H_SZ  * 2);
    uint16_t* cb_b = (uint16_t*)alloc((size_t)C_SZ  * HR_SZ * 2);
    float*    cn2  = (float*)   alloc((size_t)C_SZ * 4);
    int*      idxb = (int*)     alloc((size_t)NTOK * 4);
    float*    part = (float*)   alloc((size_t)NTOK * 4);

    // output layout: recon | quantized_st | token_indices(float) | commit_loss
    float* recon = (float*)d_out;
    float* q_st  = recon + (size_t)NTOK * H_SZ;
    float* tok   = q_st  + (size_t)NTOK * HR_SZ;
    float* loss  = tok   + NTOK;

    // 1) one-time conversions: activations/codebook -> bf16, weights -> bf16 transposed [N][K]
    cvt_f32_bf16<<<(NTOK * H_SZ) / 256, 256, 0, stream>>>(x, x_b, NTOK * H_SZ);
    cvt_f32_bf16<<<(C_SZ * HR_SZ) / 256, 256, 0, stream>>>(codebook, cb_b, C_SZ * HR_SZ);
    transpose_cvt<<<dim3(H_SZ / 32,  H_SZ / 32),  256, 0, stream>>>(enc_w1, w1t, H_SZ,  H_SZ);
    transpose_cvt<<<dim3(HR_SZ / 32, H_SZ / 32),  256, 0, stream>>>(enc_w2, w2t, H_SZ,  HR_SZ);
    transpose_cvt<<<dim3(H_SZ / 32,  HR_SZ / 32), 256, 0, stream>>>(dec_w1, w3t, HR_SZ, H_SZ);
    transpose_cvt<<<dim3(H_SZ / 32,  H_SZ / 32),  256, 0, stream>>>(dec_w2, w4t, H_SZ,  H_SZ);
    cnorm_kernel<<<C_SZ / 8, 256, 0, stream>>>(codebook, cn2);

    // 2) encoder: h = relu(x@W1+b1) ; z = h@W2+b2 (fp32 + bf16 copies of z)
    gemm_bf16_wmma<<<dim3(H_SZ / TN, NTOK / TM), 256, 0, stream>>>(
        x_b, w1t, enc_b1, nullptr, h_b, NTOK, H_SZ, H_SZ, 1);
    gemm_bf16_wmma<<<dim3(HR_SZ / TN, NTOK / TM), 256, 0, stream>>>(
        h_b, w2t, enc_b2, z_f, z_b, NTOK, HR_SZ, H_SZ, 0);

    // 3) fused nearest-codebook search (GEMM + running argmin)
    dist_argmin_wmma<<<NTOK / TM, 256, 0, stream>>>(z_b, cb_b, cn2, idxb, tok);

    // 4) gather quantized rows, commitment-loss partials, bf16 copy for decoder
    gather_commit<<<NTOK, 128, 0, stream>>>(z_f, codebook, idxb, q_st, q_b, part);
    commit_final<<<1, 256, 0, stream>>>(part, loss);

    // 5) decoder: h2 = relu(q@Wd1+bd1) ; recon = h2@Wd2+bd2 (fp32 -> d_out)
    gemm_bf16_wmma<<<dim3(H_SZ / TN, NTOK / TM), 256, 0, stream>>>(
        q_b, w3t, dec_b1, nullptr, h_b, NTOK, H_SZ, HR_SZ, 1);
    gemm_bf16_wmma<<<dim3(H_SZ / TN, NTOK / TM), 256, 0, stream>>>(
        h_b, w4t, dec_b2, recon, nullptr, NTOK, H_SZ, H_SZ, 0);
}
